// SplineCritic_47588237640103
// MI455X (gfx1250) — matrix operand order repaired and checked by reading, not compiled, not run on hardware
//
#include <hip/hip_runtime.h>
#include <hip/hip_bf16.h>

// ---------------------------------------------------------------------------
// SplineCritic on gfx1250: fused KAN spline layers via v_wmma_f32_16x16x32_bf16
//
// Layer-as-GEMM:  out = [silu(x) | rbf(x)] @ [base_w.T ; spline_w.T] + (bb+sb)
//   layer1: M=4096, K=2304, N=128      layer2: M=4096, K=1152, N=64
// A (16x32 bf16) is generated in-register from an LDS-staged x tile with
// uniform (branch-free) silu / rbf phases; rbf uses z_i = (x+2)*1.75 - i.
// silu uses v_rcp_f32 (result is rounded to bf16 anyway; no IEEE div chains).
// B is pre-packed to the exact WMMA lane layout (contiguous v16bf per lane).
// Both x/y branches run in one launch (blockIdx.y).
// ---------------------------------------------------------------------------

typedef __attribute__((ext_vector_type(16))) __bf16 v16bf;
typedef __attribute__((ext_vector_type(8)))  float  v8f;

#define INVH 1.75f            // 7/4 ; grid spacing h = 4/7, NUM_GRIDS=8 on [-2,2]

__device__ __forceinline__ float fast_silu(float x) {
  // x * sigmoid(x); v_exp + v_add + v_rcp + v_mul (all co-executable w/ XDL)
  return x * __builtin_amdgcn_rcpf(1.0f + __expf(-x));
}

// ---- weight packing: B[K][N] -> per-lane WMMA fragment order -------------
// packed index p = ((kt*NT + nt)*32 + lane)*16 + e
//   K = kt*32 + ((e>>3)<<4) + ((lane>>4)<<3) + (e&7) ,  N = nt*16 + (lane&15)
__global__ void pack_weights(const float* __restrict__ bw,   // [NOUT, K1]
                             const float* __restrict__ sw,   // [NOUT, K2]
                             __bf16* __restrict__ dst,
                             int K1, int KTOT, int NOUT) {
  int p = blockIdx.x * blockDim.x + threadIdx.x;
  int NT = NOUT >> 4;
  int total = (KTOT >> 5) * NT * 512;
  if (p >= total) return;
  int e  = p & 15;
  int l  = (p >> 4) & 31;
  int t  = p >> 9;
  int nt = t % NT;
  int kt = t / NT;
  int K  = (kt << 5) + ((e >> 3) << 4) + ((l >> 4) << 3) + (e & 7);
  int N  = nt * 16 + (l & 15);
  int K2 = KTOT - K1;
  float w = (K < K1) ? bw[(size_t)N * K1 + K]
                     : sw[(size_t)N * K2 + (K - K1)];
  dst[p] = (__bf16)w;
}

__global__ void zero_partials(float* __restrict__ p) {
  if (threadIdx.x < 8) p[threadIdx.x] = 0.0f;
}

// One wave per block; wave owns a 16-row x full-N strip. blockIdx.y = branch.
template <int DIN, int KTOT, int NOUT, bool REDUCE>
__global__ __launch_bounds__(32)
void spline_layer(const float*  __restrict__ X0, const float*  __restrict__ X1,
                  const __bf16* __restrict__ W0, const __bf16* __restrict__ W1,
                  const float*  __restrict__ bb0, const float* __restrict__ bb1,
                  const float*  __restrict__ sb0, const float* __restrict__ sb1,
                  float* __restrict__ O0, float* __restrict__ O1,
                  const float* __restrict__ fw,   // final_w [1,128] (REDUCE)
                  float* __restrict__ part) {     // partials [8]    (REDUCE)
  constexpr int NT     = NOUT / 16;
  constexpr int STRIDE = DIN + 1;                 // 257/129: conflict-free
  __shared__ float xs[16 * STRIDE];

  const int br   = blockIdx.y;
  const int tile = blockIdx.x;
  const int l    = threadIdx.x;

  const float*  X  = br ? X1 : X0;
  const __bf16* Wp = br ? W1 : W0;
  const float*  bb = br ? bb1 : bb0;
  const float*  sb = br ? sb1 : sb0;

  // stage 16 rows of X into LDS (float4 global loads)
  const float* Xt = X + (size_t)tile * 16 * DIN;
  for (int i = l; i < (16 * DIN) / 4; i += 32) {
    float4 v = ((const float4*)Xt)[i];
    int r = (i * 4) / DIN, c = (i * 4) % DIN;
    float* d = &xs[r * STRIDE + c];
    d[0] = v.x; d[1] = v.y; d[2] = v.z; d[3] = v.w;
  }
  __syncthreads();

  v8f acc[NT] = {};
  const int kg = l >> 4;                // lane-half selects K sub-range
  const int m  = l & 15;                // A row
  const float* xrow = &xs[m * STRIDE];

  auto mma_step = [&](int kt, v16bf a) {
    const __bf16* wb = Wp + (size_t)(kt * NT) * 512 + (size_t)l * 16;
#pragma unroll
    for (int nt = 0; nt < NT; ++nt) {
      v16bf b = *(const v16bf*)(wb + nt * 512);
      acc[nt] = __builtin_amdgcn_wmma_f32_16x16x32_bf16(
          false, a, false, b, (short)0, acc[nt], false, false);
    }
  };

  // ---- phase 1: silu(x) block, kt in [0, DIN/32) -- branch-free ----
  constexpr int KT_SILU = DIN / 32;
  for (int kt = 0; kt < KT_SILU; ++kt) {
    v16bf a;
    const int b0 = kt * 32 + (kg << 3);
#pragma unroll
    for (int e = 0; e < 8; ++e) {
      a[e]     = (__bf16)fast_silu(xrow[b0 + e]);
      a[e + 8] = (__bf16)fast_silu(xrow[b0 + 16 + e]);
    }
    mma_step(kt, a);
  }

  // ---- phase 2: rbf block, kt in [DIN/32, KTOT/32) -- branch-free ----
  // element e<8  -> x index d0,   grid index e
  // element e>=8 -> x index d0+2, grid index e-8
  // rbf arg: (x - g_i)*invh = (x+2)*1.75 - i
  for (int kt = KT_SILU; kt < KTOT / 32; ++kt) {
    const int d0 = ((kt * 32 - DIN) >> 3) + kg;
    float z0 = (xrow[d0]     + 2.0f) * INVH;
    float z1 = (xrow[d0 + 2] + 2.0f) * INVH;
    v16bf a;
#pragma unroll
    for (int i = 0; i < 8; ++i) {
      float za = z0 - (float)i;
      float zb = z1 - (float)i;
      a[i]     = (__bf16)__expf(-za * za);
      a[i + 8] = (__bf16)__expf(-zb * zb);
    }
    mma_step(kt, a);
  }

  if constexpr (!REDUCE) {
    float* Out = br ? O1 : O0;
    // C layout: VGPR v -> row v + (kg<<3), col = nt*16 + (l&15)
#pragma unroll
    for (int nt = 0; nt < NT; ++nt) {
      int n = nt * 16 + (l & 15);
      float bias = bb[n] + sb[n];
#pragma unroll
      for (int v = 0; v < 8; ++v) {
        int mm = v + (kg << 3);
        Out[((size_t)tile * 16 + mm) * NOUT + n] = acc[nt][v] + bias;
      }
    }
  } else {
    // Fold h2 . final_w and the mean-over-S reduction: every (row,col) of the
    // tile appears exactly once across (lane, vgpr, nt); all 16 rows belong
    // to the same batch element (1024 % 16 == 0).
    const float* wseg = fw + br * NOUT;
    float local = 0.0f;
#pragma unroll
    for (int nt = 0; nt < NT; ++nt) {
      int n = nt * 16 + (l & 15);
      float w    = wseg[n];
      float bias = bb[n] + sb[n];
#pragma unroll
      for (int v = 0; v < 8; ++v) local += (acc[nt][v] + bias) * w;
    }
    for (int off = 16; off > 0; off >>= 1)
      local += __shfl_down(local, off, 32);
    if (l == 0) atomicAdd(&part[br * 4 + (tile >> 6)], local);
  }
}

__global__ void finalize(const float* __restrict__ partial,
                         const float* __restrict__ fb,
                         float* __restrict__ out) {
  int b = threadIdx.x;
  if (b < 4) out[b] = (partial[b] + partial[4 + b]) * (1.0f / 1024.0f) + fb[0];
}

// ---------------------------------------------------------------------------
extern "C" void kernel_launch(void* const* d_in, const int* in_sizes, int n_in,
                              void* d_out, int out_size, void* d_ws, size_t ws_size,
                              hipStream_t stream) {
  const float* x    = (const float*)d_in[0];
  const float* y    = (const float*)d_in[1];
  const float* bw1x = (const float*)d_in[2];
  const float* bb1x = (const float*)d_in[3];
  const float* sw1x = (const float*)d_in[4];
  const float* sb1x = (const float*)d_in[5];
  const float* bw2x = (const float*)d_in[6];
  const float* bb2x = (const float*)d_in[7];
  const float* sw2x = (const float*)d_in[8];
  const float* sb2x = (const float*)d_in[9];
  const float* bw1y = (const float*)d_in[10];
  const float* bb1y = (const float*)d_in[11];
  const float* sw1y = (const float*)d_in[12];
  const float* sb1y = (const float*)d_in[13];
  const float* bw2y = (const float*)d_in[14];
  const float* bb2y = (const float*)d_in[15];
  const float* sw2y = (const float*)d_in[16];
  const float* sb2y = (const float*)d_in[17];
  const float* fw   = (const float*)d_in[18];
  const float* fb   = (const float*)d_in[19];
  float* out = (float*)d_out;

  // workspace layout (256B-aligned regions)
  char* ws = (char*)d_ws;
  const size_t SZ_W1 = (size_t)2304 / 32 * 8 * 512 * 2;  // 589824 B
  const size_t SZ_W2 = (size_t)1152 / 32 * 4 * 512 * 2;  // 147456 B
  __bf16* pW1x = (__bf16*)(ws);
  __bf16* pW1y = (__bf16*)(ws + SZ_W1);
  __bf16* pW2x = (__bf16*)(ws + 2 * SZ_W1);
  __bf16* pW2y = (__bf16*)(ws + 2 * SZ_W1 + SZ_W2);
  float*  h1x  = (float*) (ws + 2 * SZ_W1 + 2 * SZ_W2);
  float*  h1y  = (float*) (ws + 2 * SZ_W1 + 2 * SZ_W2 + (size_t)4096 * 128 * 4);
  float*  part = (float*) (ws + 2 * SZ_W1 + 2 * SZ_W2 + (size_t)4096 * 128 * 8);

  // 1) pack weights to WMMA fragment layout + zero partial sums
  {
    int t1 = 2304 / 32 * 8 * 512;   // 294912
    int t2 = 1152 / 32 * 4 * 512;   // 73728
    pack_weights<<<(t1 + 255) / 256, 256, 0, stream>>>(bw1x, sw1x, pW1x, 256, 2304, 128);
    pack_weights<<<(t1 + 255) / 256, 256, 0, stream>>>(bw1y, sw1y, pW1y, 256, 2304, 128);
    pack_weights<<<(t2 + 255) / 256, 256, 0, stream>>>(bw2x, sw2x, pW2x, 128, 1152, 64);
    pack_weights<<<(t2 + 255) / 256, 256, 0, stream>>>(bw2y, sw2y, pW2y, 128, 1152, 64);
    zero_partials<<<1, 32, 0, stream>>>(part);
  }

  // 2) layer 1, both branches in one launch: fused silu/rbf -> WMMA -> h1
  spline_layer<256, 2304, 128, false><<<dim3(256, 2), 32, 0, stream>>>(
      x, y, pW1x, pW1y, bb1x, bb1y, sb1x, sb1y, h1x, h1y, nullptr, nullptr);

  // 3) layer 2, both branches + fused final-dot + mean reduction
  spline_layer<128, 1152, 64, true><<<dim3(256, 2), 32, 0, stream>>>(
      h1x, h1y, pW2x, pW2y, bb2x, bb2y, sb2x, sb2y, nullptr, nullptr, fw, part);

  // 4) out[b] = (px+py)/1024 + final_b
  finalize<<<1, 4, 0, stream>>>(part, fb, out);
}
